// DoubleDefCBlock_6640019439976
// MI455X (gfx1250) — matrix operand order repaired and checked by reading, not compiled, not run on hardware
//
#include <hip/hip_runtime.h>

#define HW_   128
#define NPIX  (HW_*HW_)
#define BATCH 2
#define INC_  64
#define MIDC  128
#define OUTC_ 128
#define NPTS  9

typedef __attribute__((ext_vector_type(16))) _Float16 v16h;
typedef __attribute__((ext_vector_type(8)))  _Float16 v8h;
typedef __attribute__((ext_vector_type(8)))  float    v8f;
typedef __attribute__((ext_vector_type(4)))  unsigned int v4u;
typedef __attribute__((ext_vector_type(8)))  int v8i_;
typedef __attribute__((ext_vector_type(4)))  int v4i_;

#define TDM_OK __has_builtin(__builtin_amdgcn_tensor_load_to_lds)
#if __has_builtin(__builtin_amdgcn_s_wait_tensorcnt)
#define WAIT_TENSOR(n) __builtin_amdgcn_s_wait_tensorcnt((short)(n))
#else
#define WAIT_TENSOR(n)
#endif

// LDS B-tile row stride: 64 data halves + 8 pad halves (TDM pad: 4 DWORDs / 32 DWORDs)
#define BROW 72

// ---------------- 1x1 skip conv: d_out = W_skip * x + b_skip ----------------
__global__ void k_skip(const float* __restrict__ x, const float* __restrict__ wsk,
                       const float* __restrict__ bsk, float* __restrict__ out) {
  int pix = blockIdx.x * 256 + threadIdx.x;
  int o = blockIdx.y, b = blockIdx.z;
  float acc = bsk[o];
  const float* xb = x + (size_t)b * INC_ * NPIX + pix;
  const float* wo = wsk + o * INC_;
  for (int c = 0; c < INC_; ++c) acc = fmaf(wo[c], xb[(size_t)c * NPIX], acc);
  out[((size_t)(b * OUTC_ + o)) * NPIX + pix] = acc;
}

// ---------------- offset conv: 3x3, pad 1, 18 output channels ----------------
__global__ void k_offconv(const float* __restrict__ src, const float* __restrict__ wp,
                          const float* __restrict__ bp, float* __restrict__ off, int Cin) {
  int pix = blockIdx.x * 256 + threadIdx.x;
  int oc = blockIdx.y, b = blockIdx.z;
  int i = pix >> 7, j = pix & 127;
  float acc = bp[oc];
  const float* w = wp + (size_t)oc * Cin * 9;
  const float* s = src + (size_t)b * Cin * NPIX;
  for (int c = 0; c < Cin; ++c) {
    const float* sc  = s + (size_t)c * NPIX;
    const float* wc9 = w + c * 9;
#pragma unroll
    for (int ky = 0; ky < 3; ++ky) {
      int yy = i + ky - 1;
      if ((unsigned)yy >= (unsigned)HW_) continue;
#pragma unroll
      for (int kx = 0; kx < 3; ++kx) {
        int xx = j + kx - 1;
        if ((unsigned)xx >= (unsigned)HW_) continue;
        acc = fmaf(wc9[ky * 3 + kx], sc[yy * HW_ + xx], acc);
      }
    }
  }
  off[((size_t)(b * 2 * NPTS + oc)) * NPIX + pix] = acc;
}

// ---------------- fp32 -> fp16 weight convert ----------------
__global__ void k_cvt(const float* __restrict__ w, _Float16* __restrict__ wh, int n) {
  int i = blockIdx.x * 256 + threadIdx.x;
  if (i < n) wh[i] = (_Float16)w[i];
}

// ------- deformable bilinear gather -> K-major im2col Xcol[b][k][pix] (f16) -------
__global__ void k_gather(const float* __restrict__ src, const float* __restrict__ off,
                         _Float16* __restrict__ xcol, int Cin) {
  int pix = blockIdx.x * 256 + threadIdx.x;
  int k = blockIdx.y;               // k = c*9 + p
  int b = blockIdx.z;
  int c = k / 9, p = k % 9;
  int i = pix >> 7, j = pix & 127;
  float pni = (float)(p / 3 - 1), pnj = (float)(p % 3 - 1);
  float offY = off[((size_t)(b * 2 * NPTS + p)) * NPIX + pix];
  float offX = off[((size_t)(b * 2 * NPTS + NPTS + p)) * NPIX + pix];
  float py = offY + pni + (float)(i + 1);   // padded (130x130) coords
  float px = offX + pnj + (float)(j + 1);
  float fy = floorf(py), fx = floorf(px);
  const float mx = (float)(HW_ + 1);        // 129
  float lty = fminf(fmaxf(fy, 0.f), mx);
  float ltx = fminf(fmaxf(fx, 0.f), mx);
  float rby = fminf(fmaxf(fy + 1.f, 0.f), mx);
  float rbx = fminf(fmaxf(fx + 1.f, 0.f), mx);
  float pyc = fminf(fmaxf(py, 0.f), mx);
  float pxc = fminf(fmaxf(px, 0.f), mx);
  float g_lt = (1.f + (lty - pyc)) * (1.f + (ltx - pxc));
  float g_rb = (1.f - (rby - pyc)) * (1.f - (rbx - pxc));
  float g_lb = (1.f + (lty - pyc)) * (1.f - (rbx - pxc));
  float g_rt = (1.f - (rby - pyc)) * (1.f + (ltx - pxc));
  const float* sc = src + ((size_t)b * Cin + c) * NPIX;
  auto at = [&](float yf, float xf) -> float {
    int y = (int)yf, x = (int)xf;           // interior is [1,128]
    if (y < 1 || y > HW_ || x < 1 || x > HW_) return 0.f;
    return sc[(y - 1) * HW_ + (x - 1)];
  };
  float v = g_lt * at(lty, ltx) + g_rb * at(rby, rbx)
          + g_lb * at(lty, rbx) + g_rt * at(rby, ltx);
  size_t K = (size_t)Cin * 9;
  xcol[((size_t)b * K + (size_t)k) * NPIX + pix] = (_Float16)v;
}

// ---- TDM: DMA one 32(K) x 64(N) f16 tile of Xcol into LDS with row padding ----
__device__ __forceinline__ void tdm_load_tile(const _Float16* gsrc, unsigned lds_off) {
#if TDM_OK
  unsigned long long ga = (unsigned long long)(uintptr_t)gsrc;
  v4u g0;
  g0[0] = 1u;                                               // count=1, user mode
  g0[1] = lds_off;                                          // lds_addr
  g0[2] = (unsigned)ga;                                     // global_addr[31:0]
  g0[3] = (unsigned)((ga >> 32) & 0x01ffffffu) | (2u << 30); // global_addr[56:32] | type=2
  v8i_ g1;
  g1[0] = (1 << 16) | (1 << 20) | (4 << 22) | (3 << 25);    // data_size=2B, pad_enable,
                                                            // pad_interval=32 DW, pad=4 DW
  g1[1] = (int)(16384u << 16);   // tensor_dim0 = NPIX      (bits 79:48)
  g1[2] = (32 << 16);            // tensor_dim1 = 32 rows   (bits 111:80)
  g1[3] = (64 << 16);            // tile_dim0  = 64         (bits 127:112)
  g1[4] = 32;                    // tile_dim1  = 32, tile_dim2 = 0
  g1[5] = 16384;                 // tensor_dim0_stride = NPIX (bits 207:160)
  g1[6] = 0;                     // stride hi / tensor_dim1_stride lo
  g1[7] = 0;
  v4i_ gz4 = {0, 0, 0, 0};
  v8i_ gz8 = {0, 0, 0, 0, 0, 0, 0, 0};
  __builtin_amdgcn_tensor_load_to_lds(g0, g1, gz4, gz4, gz8, 0);
#else
  (void)gsrc; (void)lds_off;
#endif
}

// ---------------- WMMA GEMM: out[b,m,n] = sum_k Wh[m,k] * Xcol[b,k,n] ----------------
// 256-thread block = 8 waves; wave w owns M-tile w*16 (Cout==128); all waves share one
// TDM-staged 32x64 B-tile in LDS (double-buffered). 4 WMMAs per wave per K-step.
__global__ void k_gemm(const _Float16* __restrict__ Wh, const _Float16* __restrict__ Xcol,
                       float* __restrict__ out, int K, int Cout, int relu, int addOut) {
  __shared__ _Float16 Bsh[2][32 * BROW];
  const int tid  = threadIdx.x;
  const int lane = tid & 31;
  const int wid  = tid >> 5;
  const int lm   = lane & 15;
  const int hi   = lane >> 4;
  const int n0   = blockIdx.x * 64;
  const int m0   = wid * 16;
  const int b    = blockIdx.z;
  const int nsteps = K >> 5;

  const _Float16* btile0 = Xcol + ((size_t)b * K) * NPIX + n0;  // row kb+r at +(kb+r)*NPIX
  const unsigned lds0 = (unsigned)(uintptr_t)&Bsh[0][0];
  const unsigned lds1 = (unsigned)(uintptr_t)&Bsh[1][0];

#if TDM_OK
  const bool issuer = (wid == 0);
  if (issuer) tdm_load_tile(btile0, lds0);
#endif

  v8f acc[4] = {};
  // A layout: lane = row m0+lm; half-lane groups take K octets {0..7,16..23} / {8..15,24..31}
  const _Float16* arow = Wh + (size_t)(m0 + lm) * K + hi * 8;

  for (int s = 0; s < nsteps; ++s) {
    const int cur = s & 1;
#if TDM_OK
    if (issuer) {
      if (s + 1 < nsteps) {
        tdm_load_tile(btile0 + (size_t)((s + 1) << 5) * NPIX, cur ? lds0 : lds1);
        WAIT_TENSOR(1);            // current tile's DMA (in-order) is complete
      } else {
        WAIT_TENSOR(0);            // last tile: drain
      }
    }
#else
    {   // manual stage: 256 threads x 16B = 4KB tile
      int row = tid >> 3, chunk = (tid & 7) * 8;
      *(v8h*)&Bsh[cur][row * BROW + chunk] =
          *(const v8h*)(btile0 + (size_t)((s << 5) + row) * NPIX + chunk);
    }
#endif
    __syncthreads();

    const int kb = s << 5;
    v8h alo = *(const v8h*)(arow + kb);
    v8h ahi = *(const v8h*)(arow + kb + 16);
    v16h a = __builtin_shufflevector(alo, ahi, 0,1,2,3,4,5,6,7,8,9,10,11,12,13,14,15);
    __builtin_prefetch(arow + kb + 32, 0, 1);

    // B layout: lane = N-column (t*16+lm); K octets {0..7,16..23} / {8..15,24..31} per half
    const _Float16* bl = &Bsh[cur][lm];
#pragma unroll
    for (int t = 0; t < 4; ++t) {
      v16h bb;
#pragma unroll
      for (int e = 0; e < 16; ++e) {
        int kk = (e < 8) ? (hi * 8 + e) : (16 + hi * 8 + (e - 8));
        bb[e] = bl[kk * BROW + t * 16];
      }
      acc[t] = __builtin_amdgcn_wmma_f32_16x16x32_f16(
          false, a, false, bb, (short)0, acc[t], false, false);
    }
    __syncthreads();
  }

  // C/D layout: VGPR r -> M = m0 + r + 8*hi ; N = n0 + t*16 + lm
#pragma unroll
  for (int t = 0; t < 4; ++t) {
#pragma unroll
    for (int r = 0; r < 8; ++r) {
      int m = m0 + r + hi * 8;
      int n = n0 + t * 16 + lm;
      size_t idx = ((size_t)(b * Cout + m)) * NPIX + n;
      float v = acc[t][r];
      if (relu) v = fmaxf(v, 0.f);
      if (addOut) v += out[idx];
      out[idx] = v;
    }
  }
}

extern "C" void kernel_launch(void* const* d_in, const int* in_sizes, int n_in,
                              void* d_out, int out_size, void* d_ws, size_t ws_size,
                              hipStream_t stream) {
  const float* x   = (const float*)d_in[0];
  const float* wp1 = (const float*)d_in[1];
  const float* bp1 = (const float*)d_in[2];
  const float* wc1 = (const float*)d_in[3];
  const float* wp2 = (const float*)d_in[4];
  const float* bp2 = (const float*)d_in[5];
  const float* wc2 = (const float*)d_in[6];
  const float* wp3 = (const float*)d_in[7];
  const float* bp3 = (const float*)d_in[8];
  const float* wc3 = (const float*)d_in[9];
  const float* wsk = (const float*)d_in[10];
  const float* bsk = (const float*)d_in[11];
  float* out = (float*)d_out;

  // Scratch carve (256B aligned): off 2.4MB, Xcol 75.5MB, h1/h2 17MB, Wh 0.3MB
  char* wsp = (char*)d_ws;
  auto carve = [&](size_t bytes) -> char* {
    char* p = wsp;
    wsp += (bytes + 255) & ~(size_t)255;
    return p;
  };
  float*    off  = (float*)   carve((size_t)BATCH * 2 * NPTS * NPIX * 4);
  _Float16* xcol = (_Float16*)carve((size_t)BATCH * (MIDC * 9) * NPIX * 2);
  float*    h1   = (float*)   carve((size_t)BATCH * MIDC * NPIX * 4);
  float*    h2   = (float*)   carve((size_t)BATCH * OUTC_ * NPIX * 4);
  _Float16* wh   = (_Float16*)carve((size_t)OUTC_ * (MIDC * 9) * 2);

  // skip conv -> origin in d_out
  k_skip<<<dim3(NPIX / 256, OUTC_, BATCH), 256, 0, stream>>>(x, wsk, bsk, out);

  auto layer = [&](const float* src, int Cin, const float* wp, const float* bp,
                   const float* wc, int Cout, float* dst, int relu, int addOut) {
    int K = Cin * 9;
    k_offconv<<<dim3(NPIX / 256, 2 * NPTS, BATCH), 256, 0, stream>>>(src, wp, bp, off, Cin);
    k_gather <<<dim3(NPIX / 256, K, BATCH),       256, 0, stream>>>(src, off, xcol, Cin);
    k_cvt    <<<dim3((Cout * K + 255) / 256),     256, 0, stream>>>(wc, wh, Cout * K);
    k_gemm   <<<dim3(NPIX / 64, 1, BATCH),        256, 0, stream>>>(wh, xcol, dst, K, Cout, relu, addOut);
  };

  layer(x,  INC_,  wp1, bp1, wc1, MIDC,  h1,  1, 0);
  layer(h1, MIDC,  wp2, bp2, wc2, OUTC_, h2,  1, 0);
  layer(h2, OUTC_, wp3, bp3, wc3, OUTC_, out, 0, 1);  // += origin already in d_out

  (void)in_sizes; (void)n_in; (void)out_size; (void)ws_size;
}
